// ZipNeRFMLP_24721831756643
// MI455X (gfx1250) — matrix operand order repaired and checked by reading, not compiled, not run on hardware
//
#include <hip/hip_runtime.h>
#include <hip/hip_bf16.h>

typedef __attribute__((ext_vector_type(16))) __bf16 v16bf;
typedef __attribute__((ext_vector_type(8)))  __bf16 v8bf;
typedef __attribute__((ext_vector_type(8)))  float  v8f;

// ---- hash-grid static config (matches reference) ----
__constant__ int       c_res[10] = {16,32,64,128,256,512,1024,2048,4096,8192};
__constant__ long long c_off[10] = {0LL,4920LL,40864LL,315496LL,2412648LL,4509800LL,
                                    6606952LL,8704104LL,10801256LL,12898408LL};

// ---- bf16 weight blob layout (padded: K->mult 32, cols->mult 16) ----
// layer:            d1      d2      v0      v1      rgb
// fi:               40      64      155     283     128
// fo:               64      128     128     128     3
// K padded:         64      64      160     288     128
// fo padded:        64      128     128     128     16
// fragments(nt*ks): 4*2     8*2     8*5     8*9     1*4
// blob offset:      0       4096    12288   32768   69632   (total 71680 bf16)
// bias offset(f32): 0       64      192     320     448     (total 464 f32)
#define WBLOB_ELEMS 71680
#define BBLOB_ELEMS 464

// =====================================================================================
// Stage 0: convert + swizzle weights into fragment-ordered bf16 blob (runs every call).
// Fragment = 512 bf16; lane l owns elements [l*16, l*16+16): element e of lane l maps to
// (k = ks*32 + (lane>>4)*8 + (e&7) + (e>>3)*16, col = nt*16 + (lane&15)); frag index = nt*KS + ks.
// =====================================================================================
__global__ __launch_bounds__(256)
void zip_prep_kernel(const float* __restrict__ w_d1, const float* __restrict__ w_d2,
                     const float* __restrict__ w_v0, const float* __restrict__ w_v1,
                     const float* __restrict__ w_rgb,
                     const float* __restrict__ b_d1, const float* __restrict__ b_d2,
                     const float* __restrict__ b_v0, const float* __restrict__ b_v1,
                     const float* __restrict__ b_rgb,
                     __bf16* __restrict__ wblob, float* __restrict__ bblob)
{
    int tid = blockIdx.x * 256 + threadIdx.x;
    if (tid < WBLOB_ELEMS) {
        int layer;
        if      (tid < 4096)  layer = 0;
        else if (tid < 12288) layer = 1;
        else if (tid < 32768) layer = 2;
        else if (tid < 69632) layer = 3;
        else                  layer = 4;
        const int woff[5] = {0, 4096, 12288, 32768, 69632};
        const int fis[5]  = {40, 64, 155, 283, 128};
        const int fos[5]  = {64, 128, 128, 128, 3};
        const int kss[5]  = {2, 2, 5, 9, 4};
        int rel    = tid - woff[layer];
        int frag   = rel >> 9;
        int within = rel & 511;
        int lane   = within >> 4;
        int e      = within & 15;
        int KS = kss[layer];
        int nt = frag / KS;
        int ks = frag - nt * KS;
        int k   = ks * 32 + ((lane >> 4) << 3) + (e & 7) + ((e >> 3) << 4);
        int col = nt * 16 + (lane & 15);
        const float* w = (layer==0) ? w_d1 : (layer==1) ? w_d2 :
                         (layer==2) ? w_v0 : (layer==3) ? w_v1 : w_rgb;
        int fi = fis[layer], fo = fos[layer];
        float v = (k < fi && col < fo) ? w[(size_t)k * fo + col] : 0.f;
        wblob[tid] = (__bf16)v;
    } else if (tid < WBLOB_ELEMS + BBLOB_ELEMS) {
        int t = tid - WBLOB_ELEMS;
        int layer = (t < 64) ? 0 : (t < 192) ? 1 : (t < 320) ? 2 : (t < 448) ? 3 : 4;
        const int boff[5] = {0, 64, 192, 320, 448};
        const int fos[5]  = {64, 128, 128, 128, 3};
        int col = t - boff[layer];
        const float* bs = (layer==0) ? b_d1 : (layer==1) ? b_d2 :
                          (layer==2) ? b_v0 : (layer==3) ? b_v1 : b_rgb;
        bblob[t] = (col < fos[layer]) ? bs[col] : 0.f;
    }
}

// =====================================================================================
// Stage 1: contract -> hash-grid trilinear encode -> erf-weighted mean over N=64 samples
// One thread per (b, n) point-sample; 2 batch rows per 128-thread block.
// =====================================================================================
__global__ __launch_bounds__(128)
void zip_hash_kernel(const float* __restrict__ means,
                     const float* __restrict__ stds,
                     const float* __restrict__ table,
                     float* __restrict__ featsOut,   // [B,40]
                     float* __restrict__ coordOut,   // [B,3]
                     int B)
{
    __shared__ float sfeat[2][40];
    __shared__ float scoord[2][3];
    const int n  = threadIdx.x;      // sample 0..63
    const int lb = threadIdx.y;      // 0..1
    const int b  = blockIdx.x * 2 + lb;
    const bool valid = (b < B);

    if (n < 40) sfeat[lb][n]  = 0.f;
    if (n < 3)  scoord[lb][n] = 0.f;
    __syncthreads();

    if (valid) {
        const size_t p = (size_t)b * 64 + n;
        float x = means[p*3+0], y = means[p*3+1], z = means[p*3+2];
        float sd = stds[p];

        // ZipNeRF linearized contract
        float msq = fmaxf(x*x + y*y + z*z, 1.1920929e-07f);
        float zx = x, zy = y, zz = z, det13 = 1.f;
        if (msq > 1.f) {
            float m  = sqrtf(msq);
            float sc = (2.f*m - 1.f) / msq;
            zx = sc*x; zy = sc*y; zz = sc*z;
            det13 = cbrtf(sc*sc / msq);
        }
        atomicAdd(&scoord[lb][0], zx*0.5f);
        atomicAdd(&scoord[lb][1], zy*0.5f);
        atomicAdd(&scoord[lb][2], zz*0.5f);

        float sstd = sd * det13 * 0.5f;                       // stds/2 after contract
        float u0 = fminf(fmaxf(zx*0.25f + 0.5f, 0.f), 1.f);   // ((z/2)+1)/2 clipped
        float u1 = fminf(fmaxf(zy*0.25f + 0.5f, 0.f), 1.f);
        float u2 = fminf(fmaxf(zz*0.25f + 0.5f, 0.f), 1.f);

        const float4* tab4 = (const float4*)table;
        for (int l = 0; l < 10; ++l) {
            float resf = (float)c_res[l];
            float wl = erff(1.f / (2.8284271247461903f * sstd * resf)); // erf(1/sqrt(8 s^2 g^2))
            float px = u0*(resf-1.f)+0.5f;
            float py = u1*(resf-1.f)+0.5f;
            float pz = u2*(resf-1.f)+0.5f;
            float gx = floorf(px), gy = floorf(py), gz = floorf(pz);
            float fx = px-gx, fy = py-gy, fz = pz-gz;
            unsigned ix = (unsigned)(int)gx, iy = (unsigned)(int)gy, iz = (unsigned)(int)gz;
            float a0=0.f, a1=0.f, a2=0.f, a3=0.f;
            #pragma unroll
            for (int c = 0; c < 8; ++c) {
                unsigned dx=(c>>2)&1u, dy=(c>>1)&1u, dz=c&1u;
                unsigned cx=ix+dx, cy=iy+dy, cz=iz+dz;
                unsigned idx;
                if (l < 3) {                       // dense (tiled) levels
                    unsigned s = (unsigned)c_res[l] + 1u;
                    idx = cx + s*cy + s*s*cz;
                } else {                           // hashed levels, TSIZE = 2^21
                    idx = (cx ^ (cy*2654435761u) ^ (cz*805459861u)) & 0x1FFFFFu;
                }
                float4 f = tab4[(size_t)c_off[l] + idx];     // b128 gather
                float wc = (dx?fx:1.f-fx)*(dy?fy:1.f-fy)*(dz?fz:1.f-fz);
                a0 += wc*f.x; a1 += wc*f.y; a2 += wc*f.z; a3 += wc*f.w;
            }
            atomicAdd(&sfeat[lb][l*4+0], wl*a0);
            atomicAdd(&sfeat[lb][l*4+1], wl*a1);
            atomicAdd(&sfeat[lb][l*4+2], wl*a2);
            atomicAdd(&sfeat[lb][l*4+3], wl*a3);
        }
    }
    __syncthreads();
    if (valid) {
        if (n < 40) featsOut[(size_t)b*40 + n] = sfeat[lb][n] * (1.f/64.f);
        if (n < 3)  coordOut[(size_t)b*3 + n]  = scoord[lb][n] * (1.f/64.f);
    }
}

// =====================================================================================
// Stage 2: fused MLP via v_wmma_f32_16x16x32_bf16. One wave32 per 16-row tile.
// A fragments: 2x ds_load_b128 from bf16 LDS. B fragments: one 32B contiguous global
// load per lane from the pre-swizzled blob. No predicated loads anywhere.
// =====================================================================================

// A-matrix 16x32 bf16 (ISA 7.12.2): lane l holds row l&15; lanes 0-15: K 0-7 / 16-23,
// lanes 16-31: K 8-15 / 24-31 -> two contiguous 8-element (16B) runs.
__device__ __forceinline__ v16bf load_afrag_bf(const __bf16* a, int stride, int lane, int kbase)
{
    const __bf16* p = a + (lane & 15) * stride + kbase + ((lane >> 4) << 3);
    v8bf lo = *(const v8bf*)(p);
    v8bf hi = *(const v8bf*)(p + 16);
    v16bf v;
    #pragma unroll
    for (int e = 0; e < 8; ++e) { v[e] = lo[e]; v[e + 8] = hi[e]; }
    return v;
}

template<int NT, int KS, bool RELU>
__device__ __forceinline__ void gemm_bf(const __bf16* aLds, int aStride,
                                        const __bf16* __restrict__ wblob,
                                        const float* __restrict__ biasp,
                                        __bf16* oLds, int oStride, int lane)
{
    #pragma unroll
    for (int nt = 0; nt < NT; ++nt) {
        int col = (lane & 15) + nt * 16;
        float bv = biasp[col];
        v8f acc;
        #pragma unroll
        for (int r = 0; r < 8; ++r) acc[r] = bv;
        #pragma unroll
        for (int ks = 0; ks < KS; ++ks) {
            v16bf af = load_afrag_bf(aLds, aStride, lane, ks * 32);
            v16bf bf = *(const v16bf*)(wblob + (((size_t)(nt * KS + ks)) << 9) + lane * 16);
            acc = __builtin_amdgcn_wmma_f32_16x16x32_bf16(false, af, false, bf,
                                                          (short)0, acc, false, false);
        }
        int mb = (lane >> 4) << 3;   // D: VGPR r -> row mb+r, col = lane&15 (+ ntile)
        #pragma unroll
        for (int r = 0; r < 8; ++r) {
            float v = acc[r];
            if (RELU) v = fmaxf(v, 0.f);
            oLds[(mb + r) * oStride + col] = (__bf16)v;
        }
    }
}

__global__ __launch_bounds__(32)
void zip_mlp_kernel(const float* __restrict__ feats,     // [B,40] (workspace)
                    const float* __restrict__ viewdirs,  // [B,3]
                    const __bf16* __restrict__ wblob,
                    const float* __restrict__ bblob,
                    float* __restrict__ densOut,   // [B]
                    float* __restrict__ rgbOut)    // [B,3]
{
    __shared__ __align__(16) __bf16 sIn [16][64];    // feats padded 40->64
    __shared__ __align__(16) __bf16 sH1 [16][64];    // relu(density layer 1)
    __shared__ __align__(16) __bf16 sX  [16][160];   // bottleneck x (128) + dir_enc (27) + pad
    __shared__ __align__(16) __bf16 sCat[16][288];   // relu(v0) (128) + inputs copy + pad
    __shared__ __align__(16) __bf16 sH2 [16][128];   // relu(v1)

    const int lane = threadIdx.x;     // single wave32 per block
    const int rb   = blockIdx.x * 16;

    // stage feats (f32 -> bf16, zero pad)
    for (int i = lane; i < 16*64; i += 32) {
        int r = i >> 6, c = i & 63;
        sIn[r][c] = (__bf16)((c < 40) ? feats[(size_t)(rb + r)*40 + c] : 0.f);
    }
    // dir encoding -> sX[:,128:160]  (identity, sin(2^d v), cos(2^d v), zero pad)
    if (lane < 16) {
        int b = rb + lane;
        float vx = viewdirs[b*3+0], vy = viewdirs[b*3+1], vz = viewdirs[b*3+2];
        __bf16* dst = &sX[lane][128];
        dst[0] = (__bf16)vx; dst[1] = (__bf16)vy; dst[2] = (__bf16)vz;
        #pragma unroll
        for (int d = 0; d < 4; ++d) {
            float s = (float)(1 << d);
            dst[3  + d*3 + 0] = (__bf16)sinf(s*vx);
            dst[3  + d*3 + 1] = (__bf16)sinf(s*vy);
            dst[3  + d*3 + 2] = (__bf16)sinf(s*vz);
            dst[15 + d*3 + 0] = (__bf16)cosf(s*vx);
            dst[15 + d*3 + 1] = (__bf16)cosf(s*vy);
            dst[15 + d*3 + 2] = (__bf16)cosf(s*vz);
        }
        #pragma unroll
        for (int pz = 27; pz < 32; ++pz) dst[pz] = (__bf16)0.f;
    }
    __syncthreads();   // single-wave WG: barrier -> S_NOP, acts as fence

    // L1: [16,40] @ [40,64] -> relu
    gemm_bf<4, 2, true >(&sIn[0][0], 64, wblob + 0,     bblob + 0,   &sH1[0][0], 64,  lane);
    __syncthreads();
    // L2: [16,64] @ [64,128] -> bottleneck x
    gemm_bf<8, 2, false>(&sH1[0][0], 64, wblob + 4096,  bblob + 64,  &sX[0][0],  160, lane);
    __syncthreads();

    // density = softplus(x[:,0] - 1)
    if (lane < 16) {
        float raw = (float)sX[lane][0] - 1.0f;
        densOut[rb + lane] = (raw > 20.f) ? raw : log1pf(expf(raw));
    }

    // L3: inputs[16,155] @ [155,128] -> relu -> sCat[:, :128]
    gemm_bf<8, 5, true >(&sX[0][0], 160, wblob + 12288, bblob + 192, &sCat[0][0], 288, lane);
    // skip connection: copy inputs into sCat[:, 128:288] (pad cols already zero in sX)
    for (int i = lane; i < 16*160; i += 32) {
        int r = i / 160, c = i - r*160;
        sCat[r][128 + c] = sX[r][c];
    }
    __syncthreads();

    // L4: [16,283] @ [283,128] -> relu
    gemm_bf<8, 9, true >(&sCat[0][0], 288, wblob + 32768, bblob + 320, &sH2[0][0], 128, lane);
    __syncthreads();

    // L5: [16,128] @ [128,3] -> sigmoid -> padded rgb (direct from f32 accumulators)
    {
        int col = lane & 15;
        float bv = bblob[448 + col];
        v8f acc;
        #pragma unroll
        for (int r = 0; r < 8; ++r) acc[r] = bv;
        #pragma unroll
        for (int ks = 0; ks < 4; ++ks) {
            v16bf af = load_afrag_bf(&sH2[0][0], 128, lane, ks * 32);
            v16bf bf = *(const v16bf*)(wblob + 69632 + (((size_t)ks) << 9) + lane * 16);
            acc = __builtin_amdgcn_wmma_f32_16x16x32_bf16(false, af, false, bf,
                                                          (short)0, acc, false, false);
        }
        if (col < 3) {
            int mb = (lane >> 4) << 3;
            #pragma unroll
            for (int r = 0; r < 8; ++r) {
                float sg = 1.f / (1.f + expf(-acc[r]));
                rgbOut[(size_t)(rb + mb + r)*3 + col] = sg * 1.002f - 0.001f;
            }
        }
    }
}

// =====================================================================================
extern "C" void kernel_launch(void* const* d_in, const int* in_sizes, int n_in,
                              void* d_out, int out_size, void* d_ws, size_t ws_size,
                              hipStream_t stream) {
    // setup_inputs order: 0 rand, 1 means, 2 stds, 3 viewdirs, 4 table,
    // 5 w_d1, 6 b_d1, 7 w_d2, 8 b_d2, 9 w_v0, 10 b_v0, 11 w_v1, 12 b_v1, 13 w_rgb, 14 b_rgb
    const float* means = (const float*)d_in[1];
    const float* stds  = (const float*)d_in[2];
    const float* vdirs = (const float*)d_in[3];
    const float* table = (const float*)d_in[4];
    const float* w_d1  = (const float*)d_in[5];
    const float* b_d1  = (const float*)d_in[6];
    const float* w_d2  = (const float*)d_in[7];
    const float* b_d2  = (const float*)d_in[8];
    const float* w_v0  = (const float*)d_in[9];
    const float* b_v0  = (const float*)d_in[10];
    const float* w_v1  = (const float*)d_in[11];
    const float* b_v1  = (const float*)d_in[12];
    const float* w_rgb = (const float*)d_in[13];
    const float* b_rgb = (const float*)d_in[14];

    const int B = in_sizes[3] / 3;            // viewdirs [B,3] -> B = 4096

    float* out      = (float*)d_out;
    float* coordOut = out;                    // [B,3]
    float* densOut  = out + (size_t)B * 3;    // [B]
    float* rgbOut   = out + (size_t)B * 4;    // [B,3]

    // workspace layout: feats f32 [B*40] | wblob bf16 [71680] | bblob f32 [464]
    float*  featsWs = (float*)d_ws;
    __bf16* wblob   = (__bf16*)((char*)d_ws + (size_t)B * 40 * sizeof(float));
    float*  bblob   = (float*)((char*)d_ws + (size_t)B * 40 * sizeof(float)
                               + (size_t)WBLOB_ELEMS * sizeof(__bf16));

    zip_prep_kernel<<<(WBLOB_ELEMS + BBLOB_ELEMS + 255) / 256, 256, 0, stream>>>(
        w_d1, w_d2, w_v0, w_v1, w_rgb, b_d1, b_d2, b_v0, b_v1, b_rgb, wblob, bblob);

    dim3 blk1(64, 2);
    zip_hash_kernel<<<(B + 1) / 2, blk1, 0, stream>>>(means, stds, table,
                                                      featsWs, coordOut, B);

    zip_mlp_kernel<<<B / 16, 32, 0, stream>>>(featsWs, vdirs, wblob, bblob,
                                              densOut, rgbOut);
}
